// SelfAttentionHead_3384434229602
// MI455X (gfx1250) — matrix operand order repaired and checked by reading, not compiled, not run on hardware
//
#include <hip/hip_runtime.h>

// SelfAttentionHead (B=4, T=4096, EMB=768, HEAD=64), fp32 in/out.
// FlashAttention: v_wmma_f32_16x16x32_f16 for projections, QK^T and PV;
// per-workgroup K/V staging into LDS via GLOBAL_LOAD_ASYNC_TO_LDS (ASYNCcnt).
// Workspace usage: ~30.3 MB.

#define BB   4
#define TT   4096
#define EMB  768
#define HEAD 64

#define AS(n) __attribute__((address_space(n)))

#if defined(__has_builtin)
#if __has_builtin(__builtin_amdgcn_global_load_async_to_lds_b128)
#define HAVE_ASYNC_LDS 1
#endif
#endif
#ifndef HAVE_ASYNC_LDS
#define HAVE_ASYNC_LDS 0
#endif

typedef __attribute__((ext_vector_type(16))) _Float16 v16h;
typedef __attribute__((ext_vector_type(8)))  _Float16 v8h;
typedef __attribute__((ext_vector_type(4)))  _Float16 v4h;
typedef __attribute__((ext_vector_type(8)))  float    v8f;
typedef __attribute__((ext_vector_type(4)))  float    v4f;
typedef int v4i __attribute__((vector_size(16)));  // int4, matches builtin sig

__device__ __forceinline__ v8f wmma_f16(v16h a, v16h b, v8f c) {
  // D = A(16x32 f16) * B(32x16 f16) + C(16x16 f32)
  return __builtin_amdgcn_wmma_f32_16x16x32_f16(
      /*neg_a=*/false, a, /*neg_b=*/false, b,
      /*c_mod=*/(short)0, c, /*reuse_a=*/false, /*reuse_b=*/false);
}

// A-fragment (16x32, f16) from a row-major matrix.
// rowp = this lane's row (row m = lane&15 of the tile).
// elements 0..7  <-> k = kb + h*16 + (0..7)
// elements 8..15 <-> k = kb + h*16 + 8 + (0..7)
__device__ __forceinline__ v16h afrag(const _Float16* rowp, int kb, int h) {
  v8h lo = *(const v8h*)(rowp + kb + h * 16);
  v8h hi = *(const v8h*)(rowp + kb + h * 16 + 8);
  return __builtin_shufflevector(lo, hi, 0, 1, 2, 3, 4, 5, 6, 7,
                                 8, 9, 10, 11, 12, 13, 14, 15);
}

// B-fragment (32x16, f16) from column-major storage (= row-major transpose).
// rowp = this lane's column data (col n = lane&15). One contiguous 32B load.
__device__ __forceinline__ v16h bfrag(const _Float16* rowp, int kb, int h) {
  return *(const v16h*)(rowp + kb + h * 16);
}

// 16-byte global -> LDS copy; async (ASYNCcnt) when available.
__device__ __forceinline__ void copy16(const _Float16* g, _Float16* l) {
#if HAVE_ASYNC_LDS
  __builtin_amdgcn_global_load_async_to_lds_b128((AS(1) v4i*)g, (AS(3) v4i*)l,
                                                 0, 0);
#else
  *(v8h*)l = *(const v8h*)g;
#endif
}

__device__ __forceinline__ void wait_async0() {
#if HAVE_ASYNC_LDS
#if __has_builtin(__builtin_amdgcn_s_wait_asynccnt)
  __builtin_amdgcn_s_wait_asynccnt(0);
#else
  asm volatile("s_wait_asynccnt 0x0" ::: "memory");
#endif
#endif
}

// ---------------- phase 0a: x (f32) -> xh (f16) ----------------
__global__ __launch_bounds__(256) void cvt_x_kernel(const float* __restrict__ x,
                                                    _Float16* __restrict__ xh,
                                                    int n) {
  int i = (blockIdx.x * blockDim.x + threadIdx.x) * 4;
  if (i >= n) return;
  v4f v = *(const v4f*)(x + i);
  v4h o;
  o[0] = (_Float16)v[0]; o[1] = (_Float16)v[1];
  o[2] = (_Float16)v[2]; o[3] = (_Float16)v[3];
  *(v4h*)(xh + i) = o;
}

// ------- phase 0b: W (EMB x 64 f32) -> Wt (64 x EMB f16), 3 matrices -------
__global__ __launch_bounds__(256) void cvt_w_kernel(const float* __restrict__ Wq,
                                                    const float* __restrict__ Wk,
                                                    const float* __restrict__ Wv,
                                                    _Float16* __restrict__ wt) {
  int i = blockIdx.x * blockDim.x + threadIdx.x;
  if (i >= 3 * HEAD * EMB) return;
  int mat = i / (HEAD * EMB);
  int r   = i % (HEAD * EMB);
  int n   = r / EMB;
  int k   = r % EMB;
  const float* W = (mat == 0) ? Wq : (mat == 1) ? Wk : Wv;
  wt[i] = (_Float16)W[k * HEAD + n];
}

// ---------------- phase 1: projections via WMMA ----------------
// grid = (M/16/4, 3), block = 128 (4 waves). Wave handles one 16x64 tile.
// mat 0 -> Q (scaled by 1/8), mat 1 -> K (row-major f16), mat 2 -> V^T f16.
__global__ __launch_bounds__(128) void proj_kernel(const _Float16* __restrict__ xh,
                                                   const _Float16* __restrict__ wt,
                                                   _Float16* __restrict__ qh,
                                                   _Float16* __restrict__ kh,
                                                   _Float16* __restrict__ vt) {
  const int lane = threadIdx.x & 31;
  const int wave = threadIdx.x >> 5;
  const int h    = lane >> 4;
  const int nl   = lane & 15;
  const int mat  = blockIdx.y;
  const int m0   = (blockIdx.x * 4 + wave) * 16;

  const _Float16* xrow = xh + (size_t)(m0 + nl) * EMB;
  const _Float16* w    = wt + (size_t)mat * (HEAD * EMB);

  v8f acc[4] = {};
#pragma unroll 4
  for (int kb = 0; kb < EMB; kb += 32) {
    v16h a = afrag(xrow, kb, h);
#pragma unroll
    for (int g = 0; g < 4; ++g) {
      const _Float16* wrow = w + (size_t)(g * 16 + nl) * EMB;
      acc[g] = wmma_f16(a, bfrag(wrow, kb, h), acc[g]);
    }
  }

  const float scale = (mat == 0) ? 0.125f : 1.0f;  // fold 1/sqrt(64) into Q
  if (mat < 2) {
    _Float16* dst = (mat == 0) ? qh : kh;
#pragma unroll
    for (int g = 0; g < 4; ++g)
#pragma unroll
      for (int i = 0; i < 8; ++i)
        dst[(size_t)(m0 + i + h * 8) * HEAD + g * 16 + nl] =
            (_Float16)(acc[g][i] * scale);
  } else {
    // V^T: (B, 64, T); token index tg -> batch = tg/T, t = tg%T
#pragma unroll
    for (int g = 0; g < 4; ++g)
#pragma unroll
      for (int i = 0; i < 8; ++i) {
        int tg = m0 + i + h * 8;
        int b  = tg >> 12;        // T == 4096
        int t  = tg & (TT - 1);
        vt[((size_t)b * HEAD + g * 16 + nl) * TT + t] = (_Float16)acc[g][i];
      }
  }
}

// ---------------- phase 2: causal flash attention ----------------
// grid = (T/64, B), block = 128 (4 waves). Wave w owns query tile qblock+16w.
// All 4 waves walk the same 32-key steps; K (32x64) and V^T (64x32) blocks are
// staged once per workgroup into LDS with async global->LDS copies, then each
// wave computes S^T = K*Q^T (softmax stats land per-lane) and O^T = V^T * P^T
// (P transposed through a per-wave LDS tile).
__global__ __launch_bounds__(128) void fa_kernel(const _Float16* __restrict__ qh,
                                                 const _Float16* __restrict__ kh,
                                                 const _Float16* __restrict__ vt,
                                                 float* __restrict__ out) {
  __shared__ _Float16 kbuf[32 * 64];         // K block, [key][head]   (4 KB)
  __shared__ _Float16 vbuf[64 * 32];         // V^T block, [head][key] (4 KB)
  __shared__ _Float16 pbuf_all[4][16 * 32];  // per-wave P, [q][key]   (4 KB)

  const int tid    = threadIdx.x;
  const int lane   = tid & 31;
  const int wave   = tid >> 5;
  const int h      = lane >> 4;
  const int nl     = lane & 15;
  const int batch  = blockIdx.y;
  const int qblock = blockIdx.x * 64;
  const int qbase  = qblock + wave * 16;
  _Float16* pbuf   = &pbuf_all[wave][0];

  // Q as B-fragments (query = lane&15), head split into two 32-chunks
  const _Float16* qrow = qh + (size_t)(batch * TT + qbase + nl) * HEAD;
  const v16h qb0 = bfrag(qrow, 0, h);
  const v16h qb1 = bfrag(qrow, 32, h);

  const _Float16* kbat = kh + (size_t)batch * TT * HEAD;
  const _Float16* vbat = vt + (size_t)batch * HEAD * TT;

  v8f o[4] = {};
  float mrow = -__builtin_inff();
  float lrow = 0.0f;

  // steps cover keys needed by the block's last tile (qblock+48 .. +63)
  const int nsteps = (qblock + 95) >> 5;
  for (int step = 0; step < nsteps; ++step) {
    const int kb = step * 32;

    __syncthreads();  // everyone done reading previous K/V blocks
    // Stage K block: 32 contiguous rows of kbat -> 4 KB contiguous copy.
    // 256 chunks of 16B, 2 per thread.
    copy16(kbat + (size_t)kb * HEAD + tid * 8, kbuf + tid * 8);
    copy16(kbat + (size_t)kb * HEAD + (tid + 128) * 8, kbuf + (tid + 128) * 8);
    // Stage V^T block: 64 rows (stride T) x 32 keys.
    {
      int c0 = tid, c1 = tid + 128;
      copy16(vbat + (size_t)(c0 >> 2) * TT + kb + (c0 & 3) * 8, vbuf + c0 * 8);
      copy16(vbat + (size_t)(c1 >> 2) * TT + kb + (c1 & 3) * 8, vbuf + c1 * 8);
    }
    wait_async0();
    __syncthreads();  // K/V blocks visible to all waves

    if (kb <= qbase + 15) {  // wave-uniform: this step intersects our triangle
      // S^T tiles from LDS: rows = keys, cols = queries
      const _Float16* krow0 = kbuf + nl * HEAD;
      const _Float16* krow1 = kbuf + (16 + nl) * HEAD;
      v8f s0 = {}, s1 = {};
      s0 = wmma_f16(afrag(krow0, 0, h), qb0, s0);
      s0 = wmma_f16(afrag(krow0, 32, h), qb1, s0);
      s1 = wmma_f16(afrag(krow1, 0, h), qb0, s1);
      s1 = wmma_f16(afrag(krow1, 32, h), qb1, s1);

      // causal mask (wave-uniform branch; only diagonal steps)
      if (kb + 31 > qbase) {
#pragma unroll
        for (int i = 0; i < 8; ++i) {
          if (kb + h * 8 + i > qbase + nl)      s0[i] = -__builtin_inff();
          if (kb + 16 + h * 8 + i > qbase + nl) s1[i] = -__builtin_inff();
        }
      }

      // online softmax: stats per lane (query = nl, duplicated across halves)
      float smax = s0[0];
#pragma unroll
      for (int i = 0; i < 8; ++i) {
        smax = fmaxf(smax, s0[i]);
        smax = fmaxf(smax, s1[i]);
      }
      smax = fmaxf(smax, __shfl_xor(smax, 16, 32));
      const float mnew  = fmaxf(mrow, smax);
      const float alpha = __expf(mrow - mnew);

      v8f p0, p1;
      float rs = 0.0f;
#pragma unroll
      for (int i = 0; i < 8; ++i) {
        p0[i] = __expf(s0[i] - mnew);
        p1[i] = __expf(s1[i] - mnew);
        rs += p0[i] + p1[i];
      }
      rs += __shfl_xor(rs, 16, 32);
      lrow = lrow * alpha + rs;
      mrow = mnew;
#pragma unroll
      for (int g = 0; g < 4; ++g)
#pragma unroll
        for (int i = 0; i < 8; ++i) o[g][i] *= alpha;

      // stage P^T (C-layout) -> per-wave LDS as P row-major [q][key] f16
      v8h ph0, ph1;
#pragma unroll
      for (int i = 0; i < 8; ++i) {
        ph0[i] = (_Float16)p0[i];
        ph1[i] = (_Float16)p1[i];
      }
      *(v8h*)(pbuf + nl * 32 + h * 8)      = ph0;  // keys h*8..h*8+7
      *(v8h*)(pbuf + nl * 32 + 16 + h * 8) = ph1;  // keys 16+h*8..
      const v16h pb = *(const v16h*)(pbuf + nl * 32 + h * 16);  // B-frag of P

      // O^T += V^T(16x32 keys) * P^T(32x16 q), per 16-head group, from LDS
#pragma unroll
      for (int g = 0; g < 4; ++g) {
        const _Float16* vrow = vbuf + (g * 16 + nl) * 32;
        o[g] = wmma_f16(afrag(vrow, 0, h), pb, o[g]);
      }
    }
  }

  const float inv = 1.0f / lrow;
  // O^T layout: o[g][i] = O[query = nl][head = g*16 + i + h*8]
#pragma unroll
  for (int g = 0; g < 4; ++g)
#pragma unroll
    for (int i = 0; i < 8; ++i)
      out[(size_t)(batch * TT + qbase + nl) * HEAD + g * 16 + h * 8 + i] =
          o[g][i] * inv;
}

extern "C" void kernel_launch(void* const* d_in, const int* in_sizes, int n_in,
                              void* d_out, int out_size, void* d_ws, size_t ws_size,
                              hipStream_t stream) {
  // setup_inputs order: x, Wk, Wq, Wv
  const float* x  = (const float*)d_in[0];
  const float* Wk = (const float*)d_in[1];
  const float* Wq = (const float*)d_in[2];
  const float* Wv = (const float*)d_in[3];
  float* out = (float*)d_out;

  // workspace carve (needs ~30.3 MB)
  char* ws = (char*)d_ws;
  const size_t XH_BYTES = (size_t)BB * TT * EMB * sizeof(_Float16);   // 24 MiB
  const size_t WT_BYTES = (size_t)3 * HEAD * EMB * sizeof(_Float16);  // 288 KiB
  const size_t QKV_ELEMS = (size_t)BB * TT * HEAD;
  _Float16* xh = (_Float16*)ws;
  _Float16* wt = (_Float16*)(ws + XH_BYTES);
  _Float16* qh = (_Float16*)(ws + XH_BYTES + WT_BYTES);
  _Float16* kh = qh + QKV_ELEMS;
  _Float16* vt = kh + QKV_ELEMS;

  const int nx = BB * TT * EMB;  // 12,582,912 (divisible by 1024)
  cvt_x_kernel<<<dim3(nx / (256 * 4)), dim3(256), 0, stream>>>(x, xh, nx);
  cvt_w_kernel<<<dim3((3 * HEAD * EMB + 255) / 256), dim3(256), 0, stream>>>(
      Wq, Wk, Wv, wt);
  // M = B*T = 16384 -> 1024 tiles of 16 rows, 4 waves/block, 3 matrices
  proj_kernel<<<dim3(256, 3), dim3(128), 0, stream>>>(xh, wt, qh, kh, vt);
  // 64-query blocks (4 waves), per batch
  fa_kernel<<<dim3(TT / 64, BB), dim3(128), 0, stream>>>(qh, kh, vt, out);
}
